// MultiHeadAttention_49031346651197
// MI455X (gfx1250) — matrix operand order repaired and checked by reading, not compiled, not run on hardware
//
#include <hip/hip_runtime.h>
#include <hip/hip_bf16.h>
#include <cstdint>

// ---------------------------------------------------------------------------
// MI455X / gfx1250 multi-head attention, bf16 WMMA pipeline.
//   Stage 0: convert q,k,v,Wq,Wo f32 -> bf16 (once; vectorized)
//   Stage 1: Q/K/V projections, pure-bf16 WMMA GEMM (32x64 tile per wave)
//   Stage 2: flash attention; 64-key blocks cooperatively staged into LDS
//            via GLOBAL_LOAD_ASYNC_TO_LDS (ASYNCcnt) when available;
//            exp2-domain online softmax; conflict-free padded LDS.
//   Stage 3: output projection, f32 result
// ---------------------------------------------------------------------------

typedef __attribute__((ext_vector_type(16))) __bf16 v16bf;
typedef __attribute__((ext_vector_type(8)))  __bf16 v8bf;
typedef __attribute__((ext_vector_type(8)))  float  v8f;
typedef __attribute__((ext_vector_type(4)))  int    v4i;

#define WMMA_BF16(a, b, c) \
  __builtin_amdgcn_wmma_f32_16x16x32_bf16(false, (a), false, (b), (short)0, (c), false, false)

#define AS1 __attribute__((address_space(1)))
#define AS3 __attribute__((address_space(3)))

#if defined(__HIP_DEVICE_COMPILE__) && __has_builtin(__builtin_amdgcn_global_load_async_to_lds_b128)
#define HAVE_ASYNC_LDS 1
#else
#define HAVE_ASYNC_LDS 0
#endif

static constexpr int Bb = 4;        // batch
static constexpr int Ss = 2048;     // sequence
static constexpr int Dd = 1024;     // d_model
static constexpr int Hh = 16;       // heads
static constexpr int DK = 64;       // head dim
static constexpr int MM = Bb * Ss;  // 8192 flat rows
static constexpr int LSTR = 72;     // padded LDS row stride (elems): banks 36*ln%64 all distinct

__device__ __forceinline__ __bf16 f2bf(float f) {
  uint32_t u = __builtin_bit_cast(uint32_t, f);
  uint32_t r = (u + 0x7FFFu + ((u >> 16) & 1u)) >> 16;
  unsigned short us = (unsigned short)r;
  return __builtin_bit_cast(__bf16, us);
}

__device__ __forceinline__ uint32_t pack2bf(float a, float b) {
  uint32_t ua = __builtin_bit_cast(uint32_t, a);
  uint32_t ub = __builtin_bit_cast(uint32_t, b);
  uint32_t ra = (ua + 0x7FFFu + ((ua >> 16) & 1u)) >> 16;
  uint32_t rb = (ub + 0x7FFFu + ((ub >> 16) & 1u)) >> 16;
  return (ra & 0xFFFFu) | (rb << 16);
}

union V16U { v16bf v; v8bf h[2]; };

// A-fragment (16x32, bf16 row-major, per-lane row = lane&15).
// Layout: element e holds K = kbase + 16*half + e (matches B-fragment order).
// Two 16-byte loads (global_load_b128 / ds_load_b128).
__device__ __forceinline__ v16bf load_a_bf16(const __bf16* row, int kbase, int half) {
  V16U u;
  u.h[0] = *(const v8bf*)(row + kbase + 16 * half);
  u.h[1] = *(const v8bf*)(row + kbase + 8 + 16 * half);
  return u.v;
}

// 16-byte global -> LDS copy chunk: async (ASYNCcnt) when the builtin exists.
__device__ __forceinline__ void stage16(const __bf16* g, __bf16* l) {
#if HAVE_ASYNC_LDS
  __builtin_amdgcn_global_load_async_to_lds_b128(
      (AS1 v4i*)(AS1 void*)g, (AS3 v4i*)(AS3 void*)l, 0, 0);
#else
  *(uint4*)l = *(const uint4*)g;
#endif
}

__device__ __forceinline__ void stage_fence() {
#if HAVE_ASYNC_LDS
#if __has_builtin(__builtin_amdgcn_s_wait_asynccnt)
  __builtin_amdgcn_s_wait_asynccnt(0);
#else
  asm volatile("s_wait_asynccnt 0" ::: "memory");
#endif
#endif
}

// ---------------------------------------------------------------------------
// Stage 0: f32 -> bf16 conversion, 4 elements/thread, fully vectorized.
// ---------------------------------------------------------------------------
__global__ __launch_bounds__(256) void cvt_f32_bf16_kernel(
    const float* __restrict__ in, __bf16* __restrict__ out, int n4) {
  int i = blockIdx.x * 256 + threadIdx.x;
  if (i >= n4) return;
  float4 f = ((const float4*)in)[i];
  uint2 r;
  r.x = pack2bf(f.x, f.y);
  r.y = pack2bf(f.z, f.w);
  ((uint2*)out)[i] = r;
}

// ---------------------------------------------------------------------------
// Pure-bf16 GEMM: C = A(MMxD) @ W((N,K) torch-style)^T, wave tile 32x64.
// MODE 0: bf16 out, [b,h,s,dk]      (Q, K projections)
// MODE 1: bf16 out, [b,h,dk,s]      (V projection, transposed for PV stage)
// MODE 2: f32 out, row-major (MM,D) (final output projection)
// Block 128 thr (4 waves) -> 128x64 block tile; grid (MM/128, D/64).
// ---------------------------------------------------------------------------
template <int MODE>
__global__ __launch_bounds__(128) void gemm_bf16_kernel(
    const __bf16* __restrict__ A, const __bf16* __restrict__ W, void* __restrict__ outp) {
  const int lane = threadIdx.x & 31;
  const int wave = threadIdx.x >> 5;
  const int half = lane >> 4;
  const int ln   = lane & 15;
  const int m0 = blockIdx.x * 128 + wave * 32;
  const int n0 = blockIdx.y * 64;

  v8f acc[2][4] = {};
  const __bf16* ar0 = A + (size_t)(m0 + ln) * Dd;
  const __bf16* ar1 = A + (size_t)(m0 + 16 + ln) * Dd;

#pragma unroll 1
  for (int k = 0; k < Dd; k += 32) {
    __builtin_prefetch(ar0 + k + 64, 0, 1);  // global_prefetch_b8, speculative
    v16bf a0 = load_a_bf16(ar0, k, half);
    v16bf a1 = load_a_bf16(ar1, k, half);
#pragma unroll
    for (int t = 0; t < 4; ++t) {
      // B-fragment: lane col n = ln, element e -> K = k + 16*half + e (32B, aligned)
      v16bf b = *(const v16bf*)(W + (size_t)(n0 + t * 16 + ln) * Dd + k + 16 * half);
      acc[0][t] = WMMA_BF16(a0, b, acc[0][t]);
      acc[1][t] = WMMA_BF16(a1, b, acc[1][t]);
    }
  }

#pragma unroll
  for (int mt = 0; mt < 2; ++mt) {
#pragma unroll
    for (int t = 0; t < 4; ++t) {
#pragma unroll
      for (int v = 0; v < 8; ++v) {
        int r = m0 + mt * 16 + v + 8 * half;  // flat row = b*S + s
        int c = n0 + t * 16 + ln;             // model dim
        if constexpr (MODE == 2) {
          ((float*)outp)[(size_t)r * Dd + c] = acc[mt][t][v];
        } else {
          int b = r >> 11, sl = r & (Ss - 1);
          int h = c >> 6,  dk = c & (DK - 1);
          size_t idx = (MODE == 0)
              ? ((((size_t)b * Hh + h) * Ss + sl) * DK + dk)
              : ((((size_t)b * Hh + h) * DK + dk) * Ss + sl);
          ((__bf16*)outp)[idx] = f2bf(acc[mt][t][v]);
        }
      }
    }
  }
}

// ---------------------------------------------------------------------------
// Stage 2: flash attention. Block = 4 waves x 16 queries, shared (b,h).
// Per 64-key block: K tile (64x64) and V tile (64x64, key-minor) staged into
// LDS cooperatively (async-to-LDS), then consumed as WMMA B-fragments by all
// waves. Qh,Kh: [b,h,s,dk]; Vt: [b,h,dk,s]; AO: [b,s,d_model] (all bf16).
// ---------------------------------------------------------------------------
__global__ __launch_bounds__(128) void attn_kernel(
    const __bf16* __restrict__ Qh, const __bf16* __restrict__ Kh,
    const __bf16* __restrict__ Vt, __bf16* __restrict__ AO) {
  __shared__ __bf16 Kt[64 * LSTR];        // row = key-in-block, col = dk
  __shared__ __bf16 Vs[64 * LSTR];        // row = dk,           col = key-in-block
  __shared__ __bf16 Pl[4][16 * LSTR];     // per-wave P bounce (C-layout -> A-layout)

  const int tid  = threadIdx.x;
  const int lane = tid & 31;
  const int wave = tid >> 5;
  const int half = lane >> 4;
  const int ln   = lane & 15;
  const int bh = blockIdx.y;                   // b*16 + h
  const int q0 = blockIdx.x * 64 + wave * 16;

  const __bf16* Qb = Qh + ((size_t)bh * Ss + q0) * DK;
  const __bf16* Kb = Kh + (size_t)bh * Ss * DK;
  const __bf16* Vb = Vt + (size_t)bh * DK * Ss;

  const v16bf aQ0 = load_a_bf16(Qb + (size_t)ln * DK, 0, half);
  const v16bf aQ1 = load_a_bf16(Qb + (size_t)ln * DK, 32, half);

  v8f acc[4] = {};
  float mrow[8], lrow[8];
#pragma unroll
  for (int v = 0; v < 8; ++v) { mrow[v] = -__builtin_inff(); lrow[v] = 0.0f; }

  __bf16* P = Pl[wave];
  const float scale2 = 0.125f * 1.44269504088896f;  // log2(e)/sqrt(64)

#pragma unroll 1
  for (int key0 = 0; key0 < Ss; key0 += 64) {
    // ---- cooperative staging: 64 rows x 8 chunks(16B) each for K and V ----
#pragma unroll
    for (int i = 0; i < 4; ++i) {
      int c   = tid + i * 128;       // 0..511
      int row = c >> 3, ch = c & 7;  // chunk ch of row
      stage16(Kb + (size_t)(key0 + row) * DK + ch * 8, &Kt[row * LSTR + ch * 8]);
      stage16(Vb + (size_t)row * Ss + key0 + ch * 8,   &Vs[row * LSTR + ch * 8]);
    }
    stage_fence();        // s_wait_asynccnt 0 (producer-side completion)
    __syncthreads();      // all waves' tiles visible

    // ---- scores: 16x64 tile = 4 WMMA n-tiles x 2 k-steps over dk ----
    v8f sc[4];
#pragma unroll
    for (int nt = 0; nt < 4; ++nt) {
      const __bf16* kr = &Kt[(nt * 16 + ln) * LSTR + 16 * half];
      v16bf b0 = *(const v16bf*)(kr);
      v16bf b1 = *(const v16bf*)(kr + 32);
      v8f s = {};
      s = WMMA_BF16(aQ0, b0, s);
      s = WMMA_BF16(aQ1, b1, s);
      sc[nt] = s;
    }

    // ---- online softmax: rows in VGPR index, 16 keys/lane-group per tile ----
#pragma unroll
    for (int v = 0; v < 8; ++v) {
      float s0 = sc[0][v] * scale2, s1 = sc[1][v] * scale2;
      float s2 = sc[2][v] * scale2, s3 = sc[3][v] * scale2;
      float t = fmaxf(fmaxf(s0, s1), fmaxf(s2, s3));
#pragma unroll
      for (int off = 1; off < 16; off <<= 1) t = fmaxf(t, __shfl_xor(t, off, 32));
      float mn   = fmaxf(mrow[v], t);
      float corr = exp2f(mrow[v] - mn);
      float p0 = exp2f(s0 - mn), p1 = exp2f(s1 - mn);
      float p2 = exp2f(s2 - mn), p3 = exp2f(s3 - mn);
      float rs = (p0 + p1) + (p2 + p3);
#pragma unroll
      for (int off = 1; off < 16; off <<= 1) rs += __shfl_xor(rs, off, 32);
      lrow[v] = lrow[v] * corr + rs;
      mrow[v] = mn;
#pragma unroll
      for (int t4 = 0; t4 < 4; ++t4) acc[t4][v] *= corr;
      int rr = v + 8 * half;
      P[rr * LSTR + ln]      = f2bf(p0);
      P[rr * LSTR + 16 + ln] = f2bf(p1);
      P[rr * LSTR + 32 + ln] = f2bf(p2);
      P[rr * LSTR + 48 + ln] = f2bf(p3);
    }

    // ---- re-read P as two A-fragments (16B ds_load chunks, padded rows) ----
    v16bf aP0, aP1;
    {
      const __bf16* pr = &P[ln * LSTR];
      V16U u0, u1;
      u0.h[0] = *(const v8bf*)(pr + 16 * half);
      u0.h[1] = *(const v8bf*)(pr + 8 + 16 * half);
      u1.h[0] = *(const v8bf*)(pr + 32 + 16 * half);
      u1.h[1] = *(const v8bf*)(pr + 40 + 16 * half);
      aP0 = u0.v; aP1 = u1.v;
    }

    // ---- out += P (16x64) . V_block (64x64); V tile rows = dk, cols = key ----
#pragma unroll
    for (int t4 = 0; t4 < 4; ++t4) {
      const __bf16* vp = &Vs[(t4 * 16 + ln) * LSTR + 16 * half];
      v16bf bv0 = *(const v16bf*)(vp);
      v16bf bv1 = *(const v16bf*)(vp + 32);
      acc[t4] = WMMA_BF16(aP0, bv0, acc[t4]);
      acc[t4] = WMMA_BF16(aP1, bv1, acc[t4]);
    }

    __syncthreads();  // tiles fully consumed before next overwrite
  }

  // ---- epilogue: normalize, pack back to [b,s,d_model] bf16 ----
  const int b = bh >> 4, h = bh & 15;
#pragma unroll
  for (int t4 = 0; t4 < 4; ++t4) {
#pragma unroll
    for (int v = 0; v < 8; ++v) {
      int srow = q0 + v + 8 * half;
      float o = acc[t4][v] / lrow[v];
      AO[((size_t)b * Ss + srow) * Dd + h * DK + t4 * 16 + ln] = f2bf(o);
    }
  }
}

// ---------------------------------------------------------------------------
extern "C" void kernel_launch(void* const* d_in, const int* in_sizes, int n_in,
                              void* d_out, int out_size, void* d_ws, size_t ws_size,
                              hipStream_t stream) {
  const float* q  = (const float*)d_in[0];
  const float* k  = (const float*)d_in[1];
  const float* v  = (const float*)d_in[2];
  const float* Wq = (const float*)d_in[3];
  const float* Wo = (const float*)d_in[4];
  float* out = (float*)d_out;

  const size_t ACT = (size_t)MM * Dd;  // 8Mi elements
  const size_t WEL = (size_t)Dd * Dd;  // 1Mi elements

  // bf16 workspace layout (~105 MB). AO aliases Xq (consumed before attn runs).
  __bf16* Xq  = (__bf16*)d_ws;         // converted q        [MM,D]
  __bf16* Xk  = Xq  + ACT;             // converted k        [MM,D]
  __bf16* Xv  = Xk  + ACT;             // converted v        [MM,D]
  __bf16* Wqb = Xv  + ACT;             // converted Wq       [D,D]
  __bf16* Wob = Wqb + WEL;             // converted Wo       [D,D]
  __bf16* Qh  = Wob + WEL;             // Q projected        [b,h,s,dk]
  __bf16* Kh  = Qh  + ACT;             // K projected        [b,h,s,dk]
  __bf16* Vt  = Kh  + ACT;             // V projected        [b,h,dk,s]
  __bf16* AO  = Xq;                    // attention output   [b,s,d_model] (alias)

  // Stage 0: conversions
  const int actN4 = (int)(ACT / 4), wN4 = (int)(WEL / 4);
  cvt_f32_bf16_kernel<<<actN4 / 256, 256, 0, stream>>>(q,  Xq,  actN4);
  cvt_f32_bf16_kernel<<<actN4 / 256, 256, 0, stream>>>(k,  Xk,  actN4);
  cvt_f32_bf16_kernel<<<actN4 / 256, 256, 0, stream>>>(v,  Xv,  actN4);
  cvt_f32_bf16_kernel<<<wN4 / 256,   256, 0, stream>>>(Wq, Wqb, wN4);
  cvt_f32_bf16_kernel<<<wN4 / 256,   256, 0, stream>>>(Wo, Wob, wN4);

  // Stage 1: projections (reference uses Wq for Q, K and V)
  dim3 gP(MM / 128, Dd / 64);  // (64, 16)
  gemm_bf16_kernel<0><<<gP, 128, 0, stream>>>(Xq, Wqb, Qh);
  gemm_bf16_kernel<0><<<gP, 128, 0, stream>>>(Xk, Wqb, Kh);
  gemm_bf16_kernel<1><<<gP, 128, 0, stream>>>(Xv, Wqb, Vt);

  // Stage 2: attention
  attn_kernel<<<dim3(Ss / 64, Bb * Hh), 128, 0, stream>>>(Qh, Kh, Vt, AO);

  // Stage 3: output projection -> f32
  gemm_bf16_kernel<2><<<gP, 128, 0, stream>>>(AO, Wob, out);
}
